// GT_FID_30391188587301
// MI455X (gfx1250) — matrix operand order repaired and checked by reading, not compiled, not run on hardware
//
#include <hip/hip_runtime.h>
#include <hip/hip_bf16.h>

#define DEVI __device__ __forceinline__

typedef __attribute__((ext_vector_type(8)))  __bf16 bf16x8;
typedef __attribute__((ext_vector_type(16))) __bf16 bf16x16;
typedef __attribute__((ext_vector_type(8)))  float  f32x8;

// Problem constants (match reference)
constexpr int Bb = 64, Tt = 512, Ee = 128, Hh = 256, G4 = 4 * Hh;      // 1024
constexpr int Nn = 50000, NEe = 800000, Gg = 64, GHh = 128, FDd = 384, NCc = 2;

// ---------------- helpers ----------------
DEVI unsigned short f2bf_bits(float f) {
  unsigned u = __builtin_bit_cast(unsigned, f);
  unsigned r = u + 0x7FFFu + ((u >> 16) & 1u);      // round-to-nearest-even
  return (unsigned short)(r >> 16);
}
DEVI float bf2f(unsigned short h) {
  unsigned u = ((unsigned)h) << 16;
  return __builtin_bit_cast(float, u);
}
DEVI float sigmoidf_(float x) { return 1.0f / (1.0f + __expf(-x)); }

// Opaque zero offset (re-materialized per loop iteration): defeats LICM's
// hoist-and-spill of the loop-invariant weight loads, WITHOUT destroying the
// pointer's address-space derivation (keeps global_load, not flat_load).
DEVI unsigned opaque_zero() {
  unsigned z = 0;
  asm volatile("" : "+s"(z));
  return z;
}

DEVI f32x8 wmma_bf16(bf16x16 a, bf16x16 b, f32x8 c) {
  // v_wmma_f32_16x16x32_bf16  (neg_a, A, neg_b, B, c_mod, C, reuse_a, reuse_b)
  return __builtin_amdgcn_wmma_f32_16x16x32_bf16(false, a, false, b, (short)0, c,
                                                 false, false);
}
// A-operand (16x32 bf16): lane holds two 8-halfword chunks:
// k = base..base+7 and base+8..base+15 where base = kk + 16*(lane>=16)
DEVI bf16x16 load_a_pair(const unsigned short* p) {
  bf16x8 lo = *(const bf16x8*)(p);
  bf16x8 hi = *(const bf16x8*)(p + 8);
  return __builtin_shufflevector(lo, hi, 0, 1, 2, 3, 4, 5, 6, 7, 8, 9, 10, 11, 12,
                                 13, 14, 15);
}
// B-operand (32x16 bf16): lane holds 16 contiguous K at kk + 16*(lane>=16), n = lane&15
DEVI bf16x16 load_b16(const unsigned short* p) { return *(const bf16x16*)(p); }

// ---------------- generic elementwise kernels ----------------
__global__ void k_cast(const float* __restrict__ s, unsigned short* __restrict__ d, int n) {
  int i = blockIdx.x * blockDim.x + threadIdx.x;
  if (i < n) d[i] = f2bf_bits(s[i]);
}
__global__ void k_fill(float* __restrict__ p, float v, int n) {
  int i = blockIdx.x * blockDim.x + threadIdx.x;
  if (i < n) p[i] = v;
}
__global__ void k_wg2t(const float* __restrict__ W, unsigned short* __restrict__ Wt) {
  int i = blockIdx.x * blockDim.x + threadIdx.x;
  if (i < GHh * GHh) {
    int k = i / GHh, n = i % GHh;
    Wt[n * GHh + k] = f2bf_bits(W[k * GHh + n]);   // store n-major: B(k,n)=W[k,n]
  }
}
__global__ void k_embed(const int* __restrict__ seqs, const float* __restrict__ emb,
                        unsigned short* __restrict__ exb) {
  int i = blockIdx.x * blockDim.x + threadIdx.x;
  if (i < Bb * Tt * Ee) {
    int bt = i / Ee, e = i % Ee;
    int tok = seqs[bt];
    exb[i] = f2bf_bits(emb[(size_t)tok * Ee + e]);
  }
}

// ---------------- persistent BiLSTM recurrence ----------------
// Batches are independent: grid = 2 dirs x 4 batch tiles of 16 rows.
// Block = 512 threads (16 waves); wave w owns h-tile ht=w, i.e. the four
// 16x16 gate tiles n = g*256 + ht*16 for its 16 batch rows. One task per
// wave: acc(32) + cell(8) VGPRs. Cell state never leaves VGPRs; hidden
// state goes through an 8KB LDS bf16 buffer with two barriers per step.
// Weights stream from L2 every step (opaque offset defeats LICM spilling).
__global__ __launch_bounds__(512) void k_lstm(
    const unsigned short* __restrict__ exb,
    const unsigned short* __restrict__ wih_f, const unsigned short* __restrict__ whh_f,
    const float* __restrict__ bih_f, const float* __restrict__ bhh_f,
    const unsigned short* __restrict__ wih_b, const unsigned short* __restrict__ whh_b,
    const float* __restrict__ bih_b, const float* __restrict__ bhh_b,
    const int* __restrict__ seq_lens, float* __restrict__ h_lstm) {
  const int dir = blockIdx.x & 1;
  const int mblk = blockIdx.x >> 1;         // batch tile 0..3 (16 rows each)
  const int mbase = mblk * 16;
  const unsigned short* wih = dir ? wih_b : wih_f;
  const unsigned short* whh = dir ? whh_b : whh_f;
  const float* bih = dir ? bih_b : bih_f;
  const float* bhh = dir ? bhh_b : bhh_f;

  __shared__ unsigned short h_bf[16 * Hh];  // 8 KB bf16 hidden state (16 batches)
  __shared__ int lens_s[16];

  for (int i = threadIdx.x; i < 16 * Hh; i += blockDim.x) h_bf[i] = 0;
  if (threadIdx.x < 16) lens_s[threadIdx.x] = seq_lens[mbase + threadIdx.x];
  __syncthreads();

  const int lane = threadIdx.x & 31;
  const int ht = threadIdx.x >> 5;          // h-tile 0..15
  const int hiHalf = lane >> 4;
  const int l15 = lane & 15;
  const int arow = mbase + l15;             // global batch row for A operand
  const int col = ht * 16 + l15;            // hidden column this lane owns in C/D
  const int n0 = ht * 16 + l15;             // column within one gate block

  // per-gate bias, one scalar per lane, hoisted out of the recurrence
  float bv[4];
#pragma unroll
  for (int g = 0; g < 4; ++g) {
    const int n = g * Hh + n0;
    bv[g] = bih[n] + bhh[n];
  }

  f32x8 cc = {};

  for (int ts = 0; ts < Tt; ++ts) {
    const int t = dir ? (Tt - 1 - ts) : ts;
    // zero C folds to WMMA's inline-0 SRC2 => no splat/copy per step
    f32x8 acc[4] = {};

    const unsigned z = opaque_zero();       // loop-variant 0: keeps loads in loop
    const unsigned short* wihp = wih + z;   // base derivation intact -> global_load
    const unsigned short* whhp = whh + z;

    // x-part: K = E = 128, A from embedded tokens at time t
    const unsigned short* exrow = exb + ((size_t)arow * Tt + t) * Ee;
#pragma unroll
    for (int kk = 0; kk < Ee; kk += 32) {
      bf16x16 a = load_a_pair(exrow + kk + hiHalf * 16);
#pragma unroll
      for (int g = 0; g < 4; ++g) {
        const int n = g * Hh + n0;
        bf16x16 b = load_b16(wihp + (size_t)n * Ee + kk + hiHalf * 16);
        acc[g] = wmma_bf16(a, b, acc[g]);
      }
    }
    __builtin_prefetch(exrow + (dir ? -Ee : Ee), 0, 3);  // next step's token row

    // h-part: K = H = 256, A from LDS hidden state
    const unsigned short* hrow = h_bf + (size_t)l15 * Hh;
    for (int kk = 0; kk < Hh; kk += 32) {
      bf16x16 a = load_a_pair(hrow + kk + hiHalf * 16);
#pragma unroll
      for (int g = 0; g < 4; ++g) {
        const int n = g * Hh + n0;
        bf16x16 b = load_b16(whhp + (size_t)n * Hh + kk + hiHalf * 16);
        acc[g] = wmma_bf16(a, b, acc[g]);
      }
    }

    __syncthreads();  // all waves done reading h_bf

#pragma unroll
    for (int r = 0; r < 8; ++r) {
      const int row = hiHalf ? r + 8 : r;   // local batch row (C/D layout)
      const float iv = sigmoidf_(acc[0][r] + bv[0]);
      const float fv = sigmoidf_(acc[1][r] + bv[1]);
      const float gv = tanhf(acc[2][r] + bv[2]);
      const float ov = sigmoidf_(acc[3][r] + bv[3]);
      const float cn = fv * cc[r] + iv * gv;
      const float hn = ov * tanhf(cn);
      if (t < lens_s[row]) {                // masked update == packed sequences
        cc[r] = cn;
        h_bf[row * Hh + col] = f2bf_bits(hn);
      }
    }
    __syncthreads();  // h_bf visible before next step's reads
  }

  for (int i = threadIdx.x; i < 16 * Hh; i += blockDim.x) {
    int b = i / Hh, c = i % Hh;
    h_lstm[(size_t)(mbase + b) * (2 * Hh) + dir * Hh + c] = bf2f(h_bf[i]);
  }
}

// ---------------- GCN graph branch ----------------
__global__ void k_deg_acc(const int* __restrict__ dst, float* __restrict__ deg) {
  int e = blockIdx.x * blockDim.x + threadIdx.x;
  if (e < NEe) atomicAdd(&deg[dst[e]], 1.0f);
}
__global__ void k_dinv(const float* __restrict__ deg, float* __restrict__ dinv) {
  int i = blockIdx.x * blockDim.x + threadIdx.x;
  if (i < Nn) { float d = deg[i]; dinv[i] = d > 0.0f ? rsqrtf(d) : 0.0f; }
}
// layer1 is rank-1 (node_x is Nx1) => aggregate a scalar per node
__global__ void k_s1_init(const float* __restrict__ x, const float* __restrict__ dinv,
                          float* __restrict__ s1) {
  int i = blockIdx.x * blockDim.x + threadIdx.x;
  if (i < Nn) s1[i] = x[i] * dinv[i] * dinv[i];  // self-loop term
}
__global__ void k_s1_edges(const int* __restrict__ src, const int* __restrict__ dst,
                           const float* __restrict__ x, const float* __restrict__ dinv,
                           float* __restrict__ s1) {
  int e = blockIdx.x * blockDim.x + threadIdx.x;
  if (e < NEe) atomicAdd(&s1[dst[e]], x[src[e]] * dinv[src[e]] * dinv[dst[e]]);
}
__global__ void k_x1(const float* __restrict__ s1, const float* __restrict__ Wg1,
                     const float* __restrict__ bg1, float* __restrict__ x1) {
  int i = blockIdx.x * blockDim.x + threadIdx.x;
  if (i < Nn * GHh) {
    int row = i / GHh, j = i % GHh;
    x1[i] = fmaxf(s1[row] * Wg1[j] + bg1[j], 0.0f);  // relu(gcn1)
  }
}
__global__ __launch_bounds__(256) void k_bnstats(const float* __restrict__ x,
                                                 float* __restrict__ mu,
                                                 float* __restrict__ var) {
  const int j = blockIdx.x;  // one block per column
  float s = 0.0f, s2 = 0.0f;
  for (int i = threadIdx.x; i < Nn; i += 256) {
    float v = x[(size_t)i * GHh + j];
    s += v; s2 += v * v;
  }
  __shared__ float sh[256], sh2[256];
  sh[threadIdx.x] = s; sh2[threadIdx.x] = s2;
  __syncthreads();
  for (int st = 128; st > 0; st >>= 1) {
    if (threadIdx.x < st) { sh[threadIdx.x] += sh[threadIdx.x + st];
                            sh2[threadIdx.x] += sh2[threadIdx.x + st]; }
    __syncthreads();
  }
  if (threadIdx.x == 0) {
    float m = sh[0] / (float)Nn;
    mu[j] = m;
    var[j] = sh2[0] / (float)Nn - m * m;  // biased, matches BN training norm
  }
}
__global__ void k_bnapply(const float* __restrict__ x, const float* __restrict__ mu,
                          const float* __restrict__ var, const float* __restrict__ gamma,
                          const float* __restrict__ beta, unsigned short* __restrict__ xbn) {
  int i = blockIdx.x * blockDim.x + threadIdx.x;
  if (i < Nn * GHh) {
    int j = i % GHh;
    float v = (x[i] - mu[j]) * rsqrtf(var[j] + 1e-5f) * gamma[j] + beta[j];
    xbn[i] = f2bf_bits(v);
  }
}
// layer2 aggregation hoisted before the GEMM (segment_sum is linear)
__global__ void k_agg_init(const unsigned short* __restrict__ xbn,
                           const float* __restrict__ dinv, float* __restrict__ agg) {
  int i = blockIdx.x * blockDim.x + threadIdx.x;
  if (i < Nn * GHh) {
    int row = i / GHh;
    agg[i] = bf2f(xbn[i]) * dinv[row] * dinv[row];  // self loop
  }
}
__global__ void k_agg_edges(const int* __restrict__ src, const int* __restrict__ dst,
                            const unsigned short* __restrict__ xbn,
                            const float* __restrict__ dinv, float* __restrict__ agg) {
  long tid = (long)blockIdx.x * blockDim.x + threadIdx.x;
  if (tid < (long)NEe * 32) {
    int e = (int)(tid >> 5);
    int j0 = ((int)tid & 31) * 4;
    int s = src[e], d = dst[e];
    float c = dinv[s] * dinv[d];
#pragma unroll
    for (int jj = 0; jj < 4; ++jj)
      atomicAdd(&agg[(size_t)d * GHh + j0 + jj],
                bf2f(xbn[(size_t)s * GHh + j0 + jj]) * c);
  }
}
// x2 = relu(agg @ Wg2 + bg2) : 50000x128x128 WMMA GEMM, 1 tile per wave
__global__ __launch_bounds__(256) void k_gemm_gcn(const unsigned short* __restrict__ A,
                                                  const unsigned short* __restrict__ Bw,
                                                  const float* __restrict__ bias,
                                                  float* __restrict__ C) {
  const int lane = threadIdx.x & 31;
  const int wave = threadIdx.x >> 5;  // n-tile 0..7 (N = 128)
  const int hiHalf = lane >> 4;
  const int l15 = lane & 15;
  const int mrow0 = blockIdx.x * 16;
  const int n = wave * 16 + l15;

  f32x8 acc = {};
  const unsigned short* arow = A + (size_t)(mrow0 + l15) * GHh;
#pragma unroll
  for (int kk = 0; kk < GHh; kk += 32) {
    bf16x16 a = load_a_pair(arow + kk + hiHalf * 16);
    bf16x16 b = load_b16(Bw + (size_t)n * GHh + kk + hiHalf * 16);
    acc = wmma_bf16(a, b, acc);
  }
  const float bvv = bias[n];
#pragma unroll
  for (int r = 0; r < 8; ++r) {
    const int row = mrow0 + (hiHalf ? r + 8 : r);
    C[(size_t)row * GHh + n] = fmaxf(acc[r] + bvv, 0.0f);
  }
}
// per-graph mean pooling
__global__ void k_pool_cnt(const int* __restrict__ gid, float* __restrict__ cnt) {
  int i = blockIdx.x * blockDim.x + threadIdx.x;
  if (i < Nn) atomicAdd(&cnt[gid[i]], 1.0f);
}
__global__ void k_pool_acc(const int* __restrict__ gid, const float* __restrict__ x2,
                           float* __restrict__ gsum) {
  long tid = (long)blockIdx.x * blockDim.x + threadIdx.x;
  if (tid < (long)Nn * 32) {
    int i = (int)(tid >> 5);
    int j0 = ((int)tid & 31) * 4;
    int g = gid[i];
#pragma unroll
    for (int jj = 0; jj < 4; ++jj)
      atomicAdd(&gsum[(size_t)g * GHh + j0 + jj], x2[(size_t)i * GHh + j0 + jj]);
  }
}
__global__ void k_hgcn(const float* __restrict__ gsum, const float* __restrict__ cnt,
                       float* __restrict__ h) {
  int i = blockIdx.x * blockDim.x + threadIdx.x;
  if (i < Gg * GHh) { int g = i / GHh; h[i] = gsum[i] / fmaxf(cnt[g], 1.0f); }
}
// fused = [h_lstm | h_gcn] @ Wfuse^T + bfuse ; out = relu(fused) @ Wcls^T + bcls
__global__ __launch_bounds__(FDd) void k_fuse(const float* __restrict__ hl,
                                              const float* __restrict__ hg,
                                              const float* __restrict__ Wf,
                                              const float* __restrict__ bf_,
                                              const float* __restrict__ Wc,
                                              const float* __restrict__ bc,
                                              float* __restrict__ out) {
  const int b = blockIdx.x, j = threadIdx.x;
  __shared__ float rf[FDd];
  float a = bf_[j];
  const float* wr = Wf + (size_t)j * (2 * Hh + GHh);
  for (int k = 0; k < 2 * Hh; ++k) a += hl[(size_t)b * (2 * Hh) + k] * wr[k];
  for (int k = 0; k < GHh; ++k) a += hg[(size_t)b * GHh + k] * wr[2 * Hh + k];
  out[Bb * NCc + (size_t)b * FDd + j] = a;  // fused output (2nd tuple element)
  rf[j] = fmaxf(a, 0.0f);
  __syncthreads();
  if (j < NCc) {
    float o = bc[j];
    for (int k = 0; k < FDd; ++k) o += rf[k] * Wc[j * FDd + k];
    out[(size_t)b * NCc + j] = o;  // logits (1st tuple element)
  }
}

// ---------------- workspace layout ----------------
constexpr size_t A256(size_t x) { return (x + 255) & ~(size_t)255; }
constexpr size_t OFF_EXB  = 0;
constexpr size_t OFF_WIHF = OFF_EXB  + A256((size_t)Bb * Tt * Ee * 2);
constexpr size_t OFF_WIHB = OFF_WIHF + A256((size_t)G4 * Ee * 2);
constexpr size_t OFF_WHHF = OFF_WIHB + A256((size_t)G4 * Ee * 2);
constexpr size_t OFF_WHHB = OFF_WHHF + A256((size_t)G4 * Hh * 2);
constexpr size_t OFF_HL   = OFF_WHHB + A256((size_t)G4 * Hh * 2);
constexpr size_t OFF_DEG  = OFF_HL   + A256((size_t)Bb * 2 * Hh * 4);
constexpr size_t OFF_DINV = OFF_DEG  + A256((size_t)Nn * 4);
constexpr size_t OFF_S1   = OFF_DINV + A256((size_t)Nn * 4);
constexpr size_t OFF_BUFA = OFF_S1   + A256((size_t)Nn * 4);     // x1 -> aggx -> x2
constexpr size_t OFF_MU   = OFF_BUFA + A256((size_t)Nn * GHh * 4);
constexpr size_t OFF_VAR  = OFF_MU   + A256((size_t)GHh * 4);
constexpr size_t OFF_XBN  = OFF_VAR  + A256((size_t)GHh * 4);
constexpr size_t OFF_AGGB = OFF_XBN  + A256((size_t)Nn * GHh * 2);
constexpr size_t OFF_WG2T = OFF_AGGB + A256((size_t)Nn * GHh * 2);
constexpr size_t OFF_GSUM = OFF_WG2T + A256((size_t)GHh * GHh * 2);
constexpr size_t OFF_GCNT = OFF_GSUM + A256((size_t)Gg * GHh * 4);
constexpr size_t OFF_HGCN = OFF_GCNT + A256((size_t)Gg * 4);

static inline int nblk(long n, int tb) { return (int)((n + tb - 1) / tb); }

extern "C" void kernel_launch(void* const* d_in, const int* in_sizes, int n_in,
                              void* d_out, int out_size, void* d_ws, size_t ws_size,
                              hipStream_t stream) {
  (void)in_sizes; (void)n_in; (void)out_size; (void)ws_size;

  const int*   seqs   = (const int*)d_in[0];
  const int*   lens   = (const int*)d_in[1];
  const float* node_x = (const float*)d_in[2];
  const int*   eidx   = (const int*)d_in[3];
  const int*   gids   = (const int*)d_in[4];
  const float* emb    = (const float*)d_in[5];
  const float* Wih_f  = (const float*)d_in[6];
  const float* Whh_f  = (const float*)d_in[7];
  const float* bih_f  = (const float*)d_in[8];
  const float* bhh_f  = (const float*)d_in[9];
  const float* Wih_b  = (const float*)d_in[10];
  const float* Whh_b  = (const float*)d_in[11];
  const float* bih_b  = (const float*)d_in[12];
  const float* bhh_b  = (const float*)d_in[13];
  const float* Wg1    = (const float*)d_in[14];
  const float* bg1    = (const float*)d_in[15];
  const float* Wg2    = (const float*)d_in[16];
  const float* bg2    = (const float*)d_in[17];
  const float* gamma  = (const float*)d_in[18];
  const float* beta   = (const float*)d_in[19];
  const float* Wfuse  = (const float*)d_in[20];
  const float* bfuse  = (const float*)d_in[21];
  const float* Wcls   = (const float*)d_in[22];
  const float* bcls   = (const float*)d_in[23];
  float* out = (float*)d_out;

  char* ws = (char*)d_ws;
  unsigned short* exb  = (unsigned short*)(ws + OFF_EXB);
  unsigned short* wihf = (unsigned short*)(ws + OFF_WIHF);
  unsigned short* wihb = (unsigned short*)(ws + OFF_WIHB);
  unsigned short* whhf = (unsigned short*)(ws + OFF_WHHF);
  unsigned short* whhb = (unsigned short*)(ws + OFF_WHHB);
  float*          hl   = (float*)(ws + OFF_HL);
  float*          deg  = (float*)(ws + OFF_DEG);
  float*          dinv = (float*)(ws + OFF_DINV);
  float*          s1   = (float*)(ws + OFF_S1);
  float*          bufA = (float*)(ws + OFF_BUFA);
  float*          mu   = (float*)(ws + OFF_MU);
  float*          var  = (float*)(ws + OFF_VAR);
  unsigned short* xbn  = (unsigned short*)(ws + OFF_XBN);
  unsigned short* aggb = (unsigned short*)(ws + OFF_AGGB);
  unsigned short* wg2t = (unsigned short*)(ws + OFF_WG2T);
  float*          gsum = (float*)(ws + OFF_GSUM);
  float*          gcnt = (float*)(ws + OFF_GCNT);
  float*          hgcn = (float*)(ws + OFF_HGCN);

  const int* esrc = eidx;
  const int* edst = eidx + NEe;
  const int TB = 256;

  // ---- sequence branch ----
  k_cast<<<nblk(G4 * Ee, TB), TB, 0, stream>>>(Wih_f, wihf, G4 * Ee);
  k_cast<<<nblk(G4 * Ee, TB), TB, 0, stream>>>(Wih_b, wihb, G4 * Ee);
  k_cast<<<nblk(G4 * Hh, TB), TB, 0, stream>>>(Whh_f, whhf, G4 * Hh);
  k_cast<<<nblk(G4 * Hh, TB), TB, 0, stream>>>(Whh_b, whhb, G4 * Hh);
  k_embed<<<nblk((long)Bb * Tt * Ee, TB), TB, 0, stream>>>(seqs, emb, exb);
  // 2 dirs x 4 batch tiles; 512 threads = 16 waves, one gate-tile task per wave
  k_lstm<<<8, 512, 0, stream>>>(exb, wihf, whhf, bih_f, bhh_f, wihb, whhb, bih_b,
                                bhh_b, lens, hl);

  // ---- graph branch ----
  k_fill<<<nblk(Nn, TB), TB, 0, stream>>>(deg, 1.0f, Nn);  // self loop
  k_deg_acc<<<nblk(NEe, TB), TB, 0, stream>>>(edst, deg);
  k_dinv<<<nblk(Nn, TB), TB, 0, stream>>>(deg, dinv);
  k_s1_init<<<nblk(Nn, TB), TB, 0, stream>>>(node_x, dinv, s1);
  k_s1_edges<<<nblk(NEe, TB), TB, 0, stream>>>(esrc, edst, node_x, dinv, s1);
  k_x1<<<nblk((long)Nn * GHh, TB), TB, 0, stream>>>(s1, Wg1, bg1, bufA);
  k_bnstats<<<GHh, 256, 0, stream>>>(bufA, mu, var);
  k_bnapply<<<nblk((long)Nn * GHh, TB), TB, 0, stream>>>(bufA, mu, var, gamma, beta,
                                                         xbn);
  k_agg_init<<<nblk((long)Nn * GHh, TB), TB, 0, stream>>>(xbn, dinv, bufA);
  k_agg_edges<<<nblk((long)NEe * 32, TB), TB, 0, stream>>>(esrc, edst, xbn, dinv,
                                                           bufA);
  k_cast<<<nblk((long)Nn * GHh, TB), TB, 0, stream>>>(bufA, aggb, Nn * GHh);
  k_wg2t<<<nblk(GHh * GHh, TB), TB, 0, stream>>>(Wg2, wg2t);
  k_gemm_gcn<<<Nn / 16, 256, 0, stream>>>(aggb, wg2t, bg2, bufA);  // bufA := x2
  k_fill<<<nblk(Gg * GHh, TB), TB, 0, stream>>>(gsum, 0.0f, Gg * GHh);
  k_fill<<<nblk(Gg, TB), TB, 0, stream>>>(gcnt, 0.0f, Gg);
  k_pool_cnt<<<nblk(Nn, TB), TB, 0, stream>>>(gids, gcnt);
  k_pool_acc<<<nblk((long)Nn * 32, TB), TB, 0, stream>>>(gids, bufA, gsum);
  k_hgcn<<<nblk(Gg * GHh, TB), TB, 0, stream>>>(gsum, gcnt, hgcn);

  // ---- fusion + classifier ----
  k_fuse<<<Bb, FDd, 0, stream>>>(hl, hgcn, Wfuse, bfuse, Wcls, bcls, out);
}
